// NeuralBellmanFordNetwork_68513318306413
// MI455X (gfx1250) — matrix operand order repaired and checked by reading, not compiled, not run on hardware
//
#include <hip/hip_runtime.h>
#include <hip/hip_bf16.h>

#define NN 20000
#define EE 640000
#define RR 64
#define DD 32
#define BB 4
#define LL 4
#define BND (BB*NN*DD)

typedef __attribute__((ext_vector_type(16))) _Float16 v16h;
typedef __attribute__((ext_vector_type(8)))  float    v8f;

__device__ __forceinline__ v8f wmma16(v16h a, v16h b, v8f c) {
  // D = A(16x32 f16) * B(32x16 f16) + C(16x16 f32)
  return __builtin_amdgcn_wmma_f32_16x16x32_f16(false, a, false, b, (short)0, c, false, false);
}

// Native CDNA5 f32 atomic add, no return (STOREcnt path), device scope so the
// RMW is performed at a level coherent across all WGPs (agg lives in L2).
__device__ __forceinline__ void atomic_add_f32(float* p, float v) {
  asm volatile("global_atomic_add_f32 %0, %1, off scope:SCOPE_DEV"
               :: "v"(p), "v"(v) : "memory");
}

// ---------------- init ----------------
__global__ __launch_bounds__(256) void zero_k(float4* p, int n4) {
  int i = blockIdx.x * 256 + threadIdx.x;
  int stride = gridDim.x * 256;
  float4 z = make_float4(0.f, 0.f, 0.f, 0.f);
  for (; i < n4; i += stride) p[i] = z;
}

__global__ void seed_k(const int* __restrict__ source, const int* __restrict__ query_rel,
                       const float* __restrict__ query_emb,
                       float* __restrict__ hidden, float* __restrict__ boundary) {
  int tid = threadIdx.x;            // 128 threads = B * D
  int b = tid >> 5, d = tid & 31;
  int s = source[b];
  float v = query_emb[query_rel[b] * DD + d];
  int off = (b * NN + s) * DD + d;
  hidden[off]   = v;
  boundary[off] = v;
}

__global__ __launch_bounds__(256) void copy_k(float4* __restrict__ dst, const float4* __restrict__ src, int n4) {
  int i = blockIdx.x * 256 + threadIdx.x;
  int stride = gridDim.x * 256;
  for (; i < n4; i += stride) dst[i] = src[i];
}

// ---------------- edge scatter ----------------
__global__ __launch_bounds__(256)
void edge_k(const int* __restrict__ nin, const int* __restrict__ nout,
            const int* __restrict__ etype, const float* __restrict__ ew,
            const float* __restrict__ relw,      // rel_w[l] : [R][D]
            const float* __restrict__ hidden,    // [B][N][D]
            float* __restrict__ agg) {           // [B][N][D]
  __shared__ __align__(16) float sRel[RR * DD];  // 8 KB

  // Async DMA stage of rel_w[l] into LDS: 256 threads x 32 B = 8 KB.
  // INST_OFFSET is added to BOTH the LDS and global address (ISA 10.7).
  {
    unsigned lds0 = (unsigned)(uintptr_t)(&sRel[threadIdx.x * 8]);
    const float* g0 = relw + threadIdx.x * 8;
    asm volatile("global_load_async_to_lds_b128 %0, %1, off\n\t"
                 "global_load_async_to_lds_b128 %0, %1, off offset:16"
                 :: "v"(lds0), "v"(g0) : "memory");
    asm volatile("s_wait_asynccnt 0x0" ::: "memory");
  }
  __syncthreads();

  int gid = blockIdx.x * 256 + threadIdx.x;      // grid sized exactly E*8
  int e  = gid >> 3;
  int dq = (gid & 7) << 2;                       // 8 lanes * float4 = 32 = D
  int src = nin[e], dst = nout[e], et = etype[e];
  float w = ew[e];
  const float4 rv = *(const float4*)&sRel[et * DD + dq];
  float r0 = rv.x * w, r1 = rv.y * w, r2 = rv.z * w, r3 = rv.w * w;
#pragma unroll
  for (int b = 0; b < BB; ++b) {
    const float4 h = *(const float4*)&hidden[(b * NN + src) * DD + dq];
    float* ap = &agg[(b * NN + dst) * DD + dq];
    atomic_add_f32(ap + 0, h.x * r0);
    atomic_add_f32(ap + 1, h.y * r1);
    atomic_add_f32(ap + 2, h.z * r2);
    atomic_add_f32(ap + 3, h.w * r3);
  }
}

// ---------------- per-node linear + LN + ReLU + residual (WMMA) ----------------
// One wave = 16 nodes. GEMM [16 x 64] @ [64 x 32] via 4x v_wmma_f32_16x16x32_f16.
__global__ __launch_bounds__(256)
void node_k(float* __restrict__ hidden, const float* __restrict__ agg,
            const float* __restrict__ W,    // lin_w[l] : [64][32]
            const float* __restrict__ bias, // lin_b[l] : [32]
            const float* __restrict__ lng, const float* __restrict__ lnb) {
  __shared__ _Float16 sA[8][2][512];             // 8 waves x 2 K-step operands x (32 lanes * 16 halves)
  int w = threadIdx.x >> 5;
  int t = threadIdx.x & 31;
  int tile = blockIdx.x * 8 + w;                 // 5000 tiles exactly
  int bidx = tile / (NN / 16);
  int n0   = (tile % (NN / 16)) * 16;
  int rowbase = bidx * NN + n0;

  // Stage A in ISA swizzle: lane m+16*kh holds row m, K = {kh*8..kh*8+7, 16+kh*8..}
  {
    int m = t >> 1;
    int koff = (t & 1) * 32;                     // 0 -> hidden features, 32 -> agg features
    const float* srcrow = (koff == 0 ? (const float*)hidden : agg) + (rowbase + m) * DD;
    int aidx = koff ? 1 : 0;
#pragma unroll
    for (int kk = 0; kk < 32; ++kk) {
      float f = srcrow[kk];
      int kh = (kk >> 3) & 1;
      int j  = (kk & 7) | ((kk >> 4) << 3);
      sA[w][aidx][(m + (kh << 4)) * 16 + j] = (_Float16)f;
    }
  }
  __syncthreads();
  v16h a0 = *(const v16h*)&sA[w][0][t * 16];
  v16h a1 = *(const v16h*)&sA[w][1][t * 16];

  int n15 = t & 15;
  int kh  = t >> 4;
  // B layout: lane = column n, half j -> K = (j&7) + 8*kh + 16*(j>>3) (+32 for K-step 1)
  v16h b00, b01, b10, b11;
#pragma unroll
  for (int j = 0; j < 16; ++j) {
    int kl = (j & 7) + (kh << 3) + ((j >> 3) << 4);
    b00[j] = (_Float16)W[(kl)      * DD + n15];
    b01[j] = (_Float16)W[(kl)      * DD + 16 + n15];
    b10[j] = (_Float16)W[(kl + 32) * DD + n15];
    b11[j] = (_Float16)W[(kl + 32) * DD + 16 + n15];
  }

  v8f c0 = {}; v8f c1 = {};
  c0 = wmma16(a0, b00, c0);
  c0 = wmma16(a1, b10, c0);
  c1 = wmma16(a0, b01, c1);
  c1 = wmma16(a1, b11, c1);

  float bias0 = bias[n15], bias1 = bias[16 + n15];
  float g0 = lng[n15], g1 = lng[16 + n15];
  float be0 = lnb[n15], be1 = lnb[16 + n15];

#pragma unroll
  for (int r = 0; r < 8; ++r) {
    float x0 = c0[r] + bias0;
    float x1 = c1[r] + bias1;
    float s = x0 + x1;
    float q = x0 * x0 + x1 * x1;
#pragma unroll
    for (int msk = 1; msk <= 8; msk <<= 1) {     // stays within 16-lane halves
      s += __shfl_xor(s, msk, 32);
      q += __shfl_xor(q, msk, 32);
    }
    float mean = s * (1.0f / 32.0f);
    float var  = q * (1.0f / 32.0f) - mean * mean;
    float rinv = rsqrtf(var + 1e-5f);
    float y0 = fmaxf((x0 - mean) * rinv * g0 + be0, 0.0f);
    float y1 = fmaxf((x1 - mean) * rinv * g1 + be1, 0.0f);
    int row = (rowbase + r + (kh << 3)) * DD;    // C-tile: VGPR r -> row r + 8*kh, lane -> col
    hidden[row + n15]      += y0;
    hidden[row + 16 + n15] += y1;
  }
}

// ---------------- scoring MLP (WMMA) ----------------
__global__ __launch_bounds__(256)
void score_k(const float* __restrict__ hidden,
             const int* __restrict__ query_rel, const float* __restrict__ query_emb,
             const float* __restrict__ W1, const float* __restrict__ b1,
             const float* __restrict__ W2, const float* __restrict__ b2,
             float* __restrict__ out) {
  __shared__ _Float16 sA[8][2][512];
  int w = threadIdx.x >> 5;
  int t = threadIdx.x & 31;
  int tile = blockIdx.x * 8 + w;
  int bidx = tile / (NN / 16);
  int n0   = (tile % (NN / 16)) * 16;
  int rowbase = bidx * NN + n0;

  {
    int m = t >> 1;
    int koff = (t & 1) * 32;
    const float* hrow = hidden + (rowbase + m) * DD;
    const float* qrow = query_emb + query_rel[bidx] * DD;
    int aidx = koff ? 1 : 0;
#pragma unroll
    for (int kk = 0; kk < 32; ++kk) {
      float f = (koff == 0) ? hrow[kk] : qrow[kk];
      int kh = (kk >> 3) & 1;
      int j  = (kk & 7) | ((kk >> 4) << 3);
      sA[w][aidx][(m + (kh << 4)) * 16 + j] = (_Float16)f;
    }
  }
  __syncthreads();
  v16h a0 = *(const v16h*)&sA[w][0][t * 16];
  v16h a1 = *(const v16h*)&sA[w][1][t * 16];

  int n15 = t & 15;
  int kh  = t >> 4;
  v16h b00, b01, b10, b11;
#pragma unroll
  for (int j = 0; j < 16; ++j) {
    int kl = (j & 7) + (kh << 3) + ((j >> 3) << 4);
    b00[j] = (_Float16)W1[(kl)      * DD + n15];
    b01[j] = (_Float16)W1[(kl)      * DD + 16 + n15];
    b10[j] = (_Float16)W1[(kl + 32) * DD + n15];
    b11[j] = (_Float16)W1[(kl + 32) * DD + 16 + n15];
  }

  v8f c0 = {}; v8f c1 = {};
  c0 = wmma16(a0, b00, c0);
  c0 = wmma16(a1, b10, c0);
  c1 = wmma16(a0, b01, c1);
  c1 = wmma16(a1, b11, c1);

  float bias0 = b1[n15], bias1 = b1[16 + n15];
  float w2a = W2[n15], w2b = W2[16 + n15];
  float b2v = b2[0];

#pragma unroll
  for (int r = 0; r < 8; ++r) {
    float h0 = fmaxf(c0[r] + bias0, 0.0f);
    float h1 = fmaxf(c1[r] + bias1, 0.0f);
    float p = h0 * w2a + h1 * w2b;
#pragma unroll
    for (int msk = 1; msk <= 8; msk <<= 1) p += __shfl_xor(p, msk, 32);
    if (n15 == 0) out[rowbase + r + (kh << 3)] = p + b2v;  // out[b*N + node]
  }
}

// ---------------- launcher ----------------
extern "C" void kernel_launch(void* const* d_in, const int* in_sizes, int n_in,
                              void* d_out, int out_size, void* d_ws, size_t ws_size,
                              hipStream_t stream) {
  const int*   node_in   = (const int*)  d_in[0];
  const int*   node_out  = (const int*)  d_in[1];
  const int*   edge_type = (const int*)  d_in[2];
  const float* edge_w    = (const float*)d_in[3];
  const int*   source    = (const int*)  d_in[4];
  const int*   query_rel = (const int*)  d_in[5];
  const float* rel_w     = (const float*)d_in[6];   // [L][R][D]
  const float* lin_w     = (const float*)d_in[7];   // [L][64][32]
  const float* lin_b     = (const float*)d_in[8];   // [L][32]
  const float* ln_g      = (const float*)d_in[9];
  const float* ln_b      = (const float*)d_in[10];
  const float* query_emb = (const float*)d_in[11];  // [R][D]
  const float* mlp_w1    = (const float*)d_in[12];  // [64][32]
  const float* mlp_b1    = (const float*)d_in[13];
  const float* mlp_w2    = (const float*)d_in[14];  // [32][1]
  const float* mlp_b2    = (const float*)d_in[15];
  float* out = (float*)d_out;

  float* hidden   = (float*)d_ws;
  float* boundary = hidden + BND;
  float* agg      = boundary + BND;

  // hidden = boundary = 0, then seed source rows
  zero_k<<<1280, 256, 0, stream>>>((float4*)hidden, (2 * BND) / 4);
  seed_k<<<1, BB * DD, 0, stream>>>(source, query_rel, query_emb, hidden, boundary);

  const int edge_blocks = (EE * 8) / 256;     // exact: 20000
  const int node_blocks = (BB * NN / 16) / 8; // exact: 625

  for (int l = 0; l < LL; ++l) {
    copy_k<<<2500, 256, 0, stream>>>((float4*)agg, (const float4*)boundary, BND / 4);
    edge_k<<<edge_blocks, 256, 0, stream>>>(node_in, node_out, edge_type, edge_w,
                                            rel_w + l * RR * DD, hidden, agg);
    node_k<<<node_blocks, 256, 0, stream>>>(hidden, agg,
                                            lin_w + l * 64 * DD, lin_b + l * DD,
                                            ln_g + l * DD, ln_b + l * DD);
  }
  score_k<<<node_blocks, 256, 0, stream>>>(hidden, query_rel, query_emb,
                                           mlp_w1, mlp_b1, mlp_w2, mlp_b2, out);
}
